// FrustrumPointNent_v1_73933567033818
// MI455X (gfx1250) — matrix operand order repaired and checked by reading, not compiled, not run on hardware
//
#include <hip/hip_runtime.h>
#include <hip/hip_bf16.h>

#define BB   32
#define NPTS 2048
#define MPTS 512
#define OUTD 59   // 3 + 2*12 + 4*8

typedef __attribute__((ext_vector_type(16))) __bf16   v16bf;
typedef __attribute__((ext_vector_type(8)))  float    v8f;
typedef __attribute__((ext_vector_type(8)))  unsigned v8u;

// ---------------- helpers ----------------
__device__ __forceinline__ unsigned short f2bfbits(float f) {
  unsigned u = __builtin_bit_cast(unsigned, f);
  unsigned r = u + 0x7FFFu + ((u >> 16) & 1u);   // round-to-nearest-even
  return (unsigned short)(r >> 16);
}
__device__ __forceinline__ float bf2f(unsigned short h) {
  return __builtin_bit_cast(float, (unsigned)h << 16);
}
__device__ __forceinline__ float rnd01(unsigned b, unsigned m) {
  unsigned long long z = (((unsigned long long)b << 32) | m) + 0x9E3779B97F4A7C15ULL;
  z = (z ^ (z >> 30)) * 0xBF58476D1CE4E5B9ULL;
  z = (z ^ (z >> 27)) * 0x94D049BB133111EBULL;
  z ^= z >> 31;
  return (float)(z >> 40) * (1.0f / 16777216.0f);
}

// CDNA5 async copy: global (SGPR base + VGPR byte offset) -> LDS (byte addr in VGPR).
// Tracked by ASYNCcnt; completed with s_wait_asynccnt before the block barrier.
__device__ __forceinline__ void async_g2l_b32(unsigned lds_bytes, const unsigned* sbase,
                                              unsigned voff_bytes) {
  asm volatile("global_load_async_to_lds_b32 %0, %1, %2"
               :: "v"(lds_bytes), "v"(voff_bytes), "s"(sbase) : "memory");
}
__device__ __forceinline__ void wait_async0() {
  asm volatile("s_wait_asynccnt 0x0" ::: "memory");
}

// ============ WMMA batched GEMM, async-LDS staged B, 2 n-tiles per wave ============
// Wp : bf16-packed weights (O, Cp) -> dwords (O, Cp2), K-pairs in dwords
// Xp : bf16-packed activations (B, N, Cp2) dwords (channel pairs innermost)
// Y  : fp32 (B, O, N)
// block = 32*wpb threads; each wave owns a 16x32 strip; all waves share the 32-col slab.
__global__ void conv_wmma_lds(const unsigned* __restrict__ Wp,
                              const float* __restrict__ bias,
                              const unsigned* __restrict__ Xp,
                              float* __restrict__ Y,
                              int Cp2, int O, int N) {
  __shared__ unsigned tile[32 * 19];           // 32 cols x 16 K-pairs, stride 19 (bank-safe)
  const int tid    = threadIdx.x;
  const int lane   = tid & 31;
  const int wave   = tid >> 5;
  const int nwaves = blockDim.x >> 5;
  const int n0   = blockIdx.x * 32;
  const int o0   = (blockIdx.y * nwaves + wave) * 16;
  const int b    = blockIdx.z;
  const int half = lane >> 4;
  const int lan  = lane & 15;
  const int orow = o0 + lan;

  const unsigned* wrow     = Wp + (long long)orow * Cp2;
  const unsigned* xcolbase = Xp + ((long long)b * N + n0) * Cp2;

  v8f acc0 = {0.f, 0.f, 0.f, 0.f, 0.f, 0.f, 0.f, 0.f};
  v8f acc1 = {0.f, 0.f, 0.f, 0.f, 0.f, 0.f, 0.f, 0.f};

  for (int k2 = 0; k2 < Cp2; k2 += 16) {
    __syncthreads();
    for (int t = tid; t < 512; t += blockDim.x) {       // stage 32x16-dword slab
      int col = t >> 4, p = t & 15;
      unsigned ldsb = (unsigned)(unsigned long long)(&tile[col * 19 + p]);
      async_g2l_b32(ldsb, xcolbase, (unsigned)(col * Cp2 + k2 + p) * 4u);
    }
    wait_async0();
    __syncthreads();

    // A fragment: two 128-bit loads (contiguous K-pairs in packed weights)
    const uint4* wa = (const uint4*)(wrow + k2 + half * 8);
    uint4 w0 = wa[0];
    uint4 w1 = wa[1];                                   // +4 dwords
    v8u au;
    au[0] = w0.x; au[1] = w0.y; au[2] = w0.z; au[3] = w0.w;
    au[4] = w1.x; au[5] = w1.y; au[6] = w1.z; au[7] = w1.w;
    v16bf a = __builtin_bit_cast(v16bf, au);

    // B fragments from LDS: VGPR j holds K-pair (half*8 + j) for column (nt*16 + lan)
    v8u bu0, bu1;
#pragma unroll
    for (int j = 0; j < 8; ++j) bu0[j] = tile[lan * 19 + half * 8 + j];
#pragma unroll
    for (int j = 0; j < 8; ++j) bu1[j] = tile[(16 + lan) * 19 + half * 8 + j];

    acc0 = __builtin_amdgcn_wmma_f32_16x16x32_bf16(false, a, false,
             __builtin_bit_cast(v16bf, bu0), (short)0, acc0, false, false);
    acc1 = __builtin_amdgcn_wmma_f32_16x16x32_bf16(false, a, false,
             __builtin_bit_cast(v16bf, bu1), (short)0, acc1, false, false);
  }

#pragma unroll
  for (int v = 0; v < 8; ++v) {                         // D: VGPR v -> row v (+8 hi half)
    int o = o0 + v + half * 8;
    float bsv = bias[o];
    Y[((long long)b * O + o) * N + n0 + lan]      = acc0[v] + bsv;
    Y[((long long)b * O + o) * N + n0 + 16 + lan] = acc1[v] + bsv;
  }
}

// ---------------- weight / input packing ----------------
__global__ void pack_w(const float* __restrict__ W, unsigned short* __restrict__ Wp,
                       int C, int Cp, int O) {
  int idx = blockIdx.x * blockDim.x + threadIdx.x;
  if (idx >= O * Cp) return;
  int o = idx / Cp, c = idx % Cp;
  Wp[idx] = (c < C) ? f2bfbits(W[(long long)o * C + c]) : (unsigned short)0;
}

__global__ void pack_pc(const float* __restrict__ pc, unsigned short* __restrict__ XP) {
  int idx = blockIdx.x * blockDim.x + threadIdx.x;   // over B*N
  if (idx >= BB * NPTS) return;
  const float* p = pc + (long long)idx * 4;
  unsigned short* out = XP + (long long)idx * 32;    // Cp=32
  for (int c = 0; c < 32; ++c) out[c] = (c < 4) ? f2bfbits(p[c]) : (unsigned short)0;
}

// ---------------- BN (over b,n per channel) ----------------
__global__ void bn_stats_conv(const float* __restrict__ Y, float* __restrict__ ms,
                              int O, int N) {
  int o = blockIdx.x, t = threadIdx.x;
  __shared__ float ss[256], sq[256];
  float a = 0.f, q = 0.f;
  long long tot = (long long)BB * N;
  for (long long i = t; i < tot; i += 256) {
    long long bb = i / N, n = i % N;
    float v = Y[(bb * O + o) * (long long)N + n];
    a += v; q += v * v;
  }
  ss[t] = a; sq[t] = q; __syncthreads();
  for (int s = 128; s > 0; s >>= 1) {
    if (t < s) { ss[t] += ss[t + s]; sq[t] += sq[t + s]; }
    __syncthreads();
  }
  if (t == 0) {
    float inv  = 1.0f / (float)tot;
    float mean = ss[0] * inv;
    float var  = sq[0] * inv - mean * mean;
    ms[2 * o]     = mean;
    ms[2 * o + 1] = rsqrtf(var + 1e-5f);
  }
}

// normalize(+relu) and emit bf16-packed (b, n, O) u16 = next layer's input
__global__ void bn_apply_pack(const float* __restrict__ Y, const float* __restrict__ ms,
                              unsigned short* __restrict__ OutP, int O, int N, int relu) {
  long long tot = (long long)BB * N * O;
  long long idx = (long long)blockIdx.x * blockDim.x + threadIdx.x;
  if (idx >= tot) return;
  int o = (int)(idx % O);
  long long bn = idx / O;
  int n = (int)(bn % N);
  int b = (int)(bn / N);
  float v = (Y[((long long)b * O + o) * N + n] - ms[2 * o]) * ms[2 * o + 1];
  if (relu) v = fmaxf(v, 0.f);
  OutP[idx] = f2bfbits(v);
}

// ---------------- packed consumers ----------------
__global__ void maxpool_packed(const unsigned short* __restrict__ Xp, float* __restrict__ g,
                               int O, int N) {
  int idx = blockIdx.x * blockDim.x + threadIdx.x;
  if (idx >= BB * O) return;
  int b = idx / O, o = idx % O;
  const unsigned short* base = Xp + (long long)b * N * O + o;
  float m = bf2f(base[0]);
  for (int n = 1; n < N; ++n) m = fmaxf(m, bf2f(base[(long long)n * O]));
  g[idx] = m;
}

__global__ void conv2_packed(const float* __restrict__ W, const float* __restrict__ bias,
                             const unsigned short* __restrict__ Xp, float* __restrict__ Y,
                             int Cin, int O, int N) {
  long long idx = (long long)blockIdx.x * blockDim.x + threadIdx.x;
  if (idx >= (long long)BB * O * N) return;
  int n = (int)(idx % N);
  int o = (int)((idx / N) % O);
  int b = (int)(idx / ((long long)N * O));
  const unsigned short* x = Xp + ((long long)b * N + n) * Cin;
  float s = bias[o];
  for (int c = 0; c < Cin; ++c) s += W[o * Cin + c] * bf2f(x[c]);
  Y[idx] = s;
}

// feat = [broadcast(one_hot||g) (1027) || pointfeat (64)] -> packed (b,n,1120) u16
__global__ void build_feat(const float* __restrict__ GC, const unsigned short* __restrict__ PFP,
                           unsigned short* __restrict__ FEATP) {
  long long tot = (long long)BB * NPTS * 1120;
  long long idx = (long long)blockIdx.x * blockDim.x + threadIdx.x;
  if (idx >= tot) return;
  int c = (int)(idx % 1120);
  long long bn = idx / 1120;
  int n = (int)(bn % NPTS);
  int b = (int)(bn / NPTS);
  unsigned short v = 0;
  if (c < 1027)      v = f2bfbits(GC[b * 1027 + c]);
  else if (c < 1091) v = PFP[((long long)b * NPTS + n) * 64 + (c - 1027)];
  FEATP[idx] = v;
}

// ---------------- small glue ----------------
__global__ void concat_pre(const float* __restrict__ a, int la,
                           const float* __restrict__ c, int lc, float* __restrict__ out) {
  int idx = blockIdx.x * blockDim.x + threadIdx.x;
  int L = la + lc;
  if (idx >= BB * L) return;
  int b = idx / L, j = idx % L;
  out[idx] = (j < la) ? a[b * la + j] : c[b * lc + (j - la)];
}

__global__ void fc_kernel(const float* __restrict__ X, const float* __restrict__ W,
                          const float* __restrict__ bias, float* __restrict__ Y,
                          int Cin, int Of) {
  int idx = blockIdx.x * blockDim.x + threadIdx.x;
  if (idx >= BB * Of) return;
  int b = idx / Of, o = idx % Of;
  const float* x = X + (long long)b * Cin;
  const float* w = W + (long long)o * Cin;
  float s = bias[o];
  for (int c = 0; c < Cin; ++c) s += x[c] * w[c];
  Y[idx] = s;
}

__global__ void bn_fc(float* __restrict__ Y, int F, int relu) {
  int f = blockIdx.x * blockDim.x + threadIdx.x;
  if (f >= F) return;
  float s = 0.f, q = 0.f;
  for (int b = 0; b < BB; ++b) { float v = Y[b * F + f]; s += v; q += v * v; }
  float mean = s / BB;
  float var  = q / BB - mean * mean;
  float rs   = rsqrtf(var + 1e-5f);
  for (int b = 0; b < BB; ++b) {
    float v = (Y[b * F + f] - mean) * rs;
    if (relu) v = fmaxf(v, 0.f);
    Y[b * F + f] = v;
  }
}

__global__ void mask_reduce(const float* __restrict__ logits, const float* __restrict__ pc,
                            int* __restrict__ maskb, int* __restrict__ cntb,
                            float* __restrict__ meanb) {
  int b = blockIdx.x, t = threadIdx.x;
  __shared__ float sx[256], sy[256], sz[256];
  __shared__ int   sc[256];
  float ax = 0.f, ay = 0.f, az = 0.f; int ac = 0;
  for (int n = t; n < NPTS; n += 256) {
    float l0 = logits[((long long)b * 2 + 0) * NPTS + n];
    float l1 = logits[((long long)b * 2 + 1) * NPTS + n];
    int m = l0 < l1;
    maskb[b * NPTS + n] = m;
    if (m) {
      const float* p = pc + ((long long)b * NPTS + n) * 4;
      ax += p[0]; ay += p[1]; az += p[2]; ++ac;
    }
  }
  sx[t] = ax; sy[t] = ay; sz[t] = az; sc[t] = ac; __syncthreads();
  for (int s = 128; s > 0; s >>= 1) {
    if (t < s) { sx[t] += sx[t+s]; sy[t] += sy[t+s]; sz[t] += sz[t+s]; sc[t] += sc[t+s]; }
    __syncthreads();
  }
  if (t == 0) {
    int c = sc[0]; cntb[b] = c;
    float d = (float)((c > 0) ? c : 1);
    meanb[b * 3 + 0] = sx[0] / d;
    meanb[b * 3 + 1] = sy[0] / d;
    meanb[b * 3 + 2] = sz[0] / d;
  }
}

// stable compaction + random resample; emits fp32 TX (B,3,M) and packed TXP (B,M,32)
__global__ void sample_gather(const int* __restrict__ maskb, const int* __restrict__ cntb,
                              const float* __restrict__ meanb, const float* __restrict__ pc,
                              float* __restrict__ tx, unsigned short* __restrict__ txp) {
  int b = blockIdx.x, t = threadIdx.x;
  __shared__ int order[NPTS];
  if (t == 0) {
    int w = 0;
    for (int n = 0; n < NPTS; ++n) if (maskb[b * NPTS + n])  order[w++] = n;
    for (int n = 0; n < NPTS; ++n) if (!maskb[b * NPTS + n]) order[w++] = n;
  }
  __syncthreads();
  int c = cntb[b];
  for (int m = t; m < MPTS; m += 256) {
    float v0 = 0.f, v1 = 0.f, v2 = 0.f;
    if (c > 0) {
      float u = rnd01((unsigned)b, (unsigned)m);
      int pos = (int)(u * (float)c);
      if (pos > c - 1) pos = c - 1;
      if (pos < 0) pos = 0;
      int idx = order[pos];
      const float* p = pc + ((long long)b * NPTS + idx) * 4;
      v0 = p[0] - meanb[b * 3 + 0];
      v1 = p[1] - meanb[b * 3 + 1];
      v2 = p[2] - meanb[b * 3 + 2];
    }
    tx[((long long)b * 3 + 0) * MPTS + m] = v0;
    tx[((long long)b * 3 + 1) * MPTS + m] = v1;
    tx[((long long)b * 3 + 2) * MPTS + m] = v2;
    unsigned short* o = txp + ((long long)b * MPTS + m) * 32;
    o[0] = f2bfbits(v0); o[1] = f2bfbits(v1); o[2] = f2bfbits(v2);
    for (int j = 3; j < 32; ++j) o[j] = 0;
  }
}

__global__ void sub_center_pack(const float* __restrict__ tx, const float* __restrict__ cd,
                                unsigned short* __restrict__ bxp) {
  int idx = blockIdx.x * blockDim.x + threadIdx.x;   // over B*M
  if (idx >= BB * MPTS) return;
  int b = idx / MPTS, m = idx % MPTS;
  unsigned short* o = bxp + (long long)idx * 32;
  for (int cdim = 0; cdim < 3; ++cdim)
    o[cdim] = f2bfbits(tx[((long long)b * 3 + cdim) * MPTS + m] - cd[b * 3 + cdim]);
  for (int j = 3; j < 32; ++j) o[j] = 0;
}

__global__ void add_center(const float* __restrict__ cd, const float* __restrict__ meanb,
                           float* __restrict__ s1c) {
  int i = threadIdx.x;
  if (i < BB * 3) s1c[i] = cd[i] + meanb[i];
}

__global__ void assemble_out(const float* __restrict__ logits, const float* __restrict__ box,
                             const float* __restrict__ s1c, float* __restrict__ out) {
  const int per = 2 * NPTS + OUTD + 3;               // 4158
  long long idx = (long long)blockIdx.x * blockDim.x + threadIdx.x;
  if (idx >= (long long)BB * per) return;
  int b = (int)(idx / per);
  int j = (int)(idx % per);
  float v;
  if (j < 2 * NPTS) {
    int n = j >> 1, k = j & 1;
    v = logits[((long long)b * 2 + k) * NPTS + n];
  } else if (j < 2 * NPTS + OUTD) {
    v = box[b * OUTD + (j - 2 * NPTS)];
  } else {
    int c = j - 2 * NPTS - OUTD;
    v = box[b * OUTD + c] + s1c[b * 3 + c];
  }
  out[idx] = v;
}

// ---------------- host orchestration ----------------
extern "C" void kernel_launch(void* const* d_in, const int* in_sizes, int n_in,
                              void* d_out, int out_size, void* d_ws, size_t ws_size,
                              hipStream_t stream) {
  (void)in_sizes; (void)n_in; (void)out_size;
  const float* pc    = (const float*)d_in[0];
  const float* oneh  = (const float*)d_in[1];
  const float* sf_w1 = (const float*)d_in[2];  const float* sf_b1 = (const float*)d_in[3];
  const float* sf_w2 = (const float*)d_in[4];  const float* sf_b2 = (const float*)d_in[5];
  const float* sf_w3 = (const float*)d_in[6];  const float* sf_b3 = (const float*)d_in[7];
  const float* sf_w4 = (const float*)d_in[8];  const float* sf_b4 = (const float*)d_in[9];
  const float* sf_w5 = (const float*)d_in[10]; const float* sf_b5 = (const float*)d_in[11];
  const float* sh_w1 = (const float*)d_in[12]; const float* sh_b1 = (const float*)d_in[13];
  const float* sh_w2 = (const float*)d_in[14]; const float* sh_b2 = (const float*)d_in[15];
  const float* sh_w3 = (const float*)d_in[16]; const float* sh_b3 = (const float*)d_in[17];
  const float* sh_w4 = (const float*)d_in[18]; const float* sh_b4 = (const float*)d_in[19];
  const float* sh_w5 = (const float*)d_in[20]; const float* sh_b5 = (const float*)d_in[21];
  const float* t_w1  = (const float*)d_in[22]; const float* t_b1  = (const float*)d_in[23];
  const float* t_w2  = (const float*)d_in[24]; const float* t_b2  = (const float*)d_in[25];
  const float* t_w3  = (const float*)d_in[26]; const float* t_b3  = (const float*)d_in[27];
  const float* t_fw1 = (const float*)d_in[28]; const float* t_fb1 = (const float*)d_in[29];
  const float* t_fw2 = (const float*)d_in[30]; const float* t_fb2 = (const float*)d_in[31];
  const float* t_fw3 = (const float*)d_in[32]; const float* t_fb3 = (const float*)d_in[33];
  const float* b_w1  = (const float*)d_in[34]; const float* b_b1  = (const float*)d_in[35];
  const float* b_w2  = (const float*)d_in[36]; const float* b_b2  = (const float*)d_in[37];
  const float* b_w3  = (const float*)d_in[38]; const float* b_b3  = (const float*)d_in[39];
  const float* b_w4  = (const float*)d_in[40]; const float* b_b4  = (const float*)d_in[41];
  const float* b_fw1 = (const float*)d_in[42]; const float* b_fb1 = (const float*)d_in[43];
  const float* b_fw2 = (const float*)d_in[44]; const float* b_fb2 = (const float*)d_in[45];
  const float* b_fw3 = (const float*)d_in[46]; const float* b_fb3 = (const float*)d_in[47];

  char* ws = (char*)d_ws;
  size_t off = 0;
  auto alloc = [&](size_t bytes) -> char* {
    char* p = ws + off;
    off += (bytes + 255) & ~(size_t)255;
    return p;
  };
  float*          YF    = (float*)alloc((size_t)BB * 1024 * NPTS * 4);   // fp32 GEMM out
  unsigned short* AP0   = (unsigned short*)alloc((size_t)BB * NPTS * 1024 * 2);
  unsigned short* AP1   = (unsigned short*)alloc((size_t)BB * NPTS * 1024 * 2);
  unsigned short* FEATP = (unsigned short*)alloc((size_t)BB * NPTS * 1120 * 2);
  unsigned short* PFP   = (unsigned short*)alloc((size_t)BB * NPTS * 64 * 2);
  unsigned short* XP    = (unsigned short*)alloc((size_t)BB * NPTS * 32 * 2);
  unsigned short* TXP   = (unsigned short*)alloc((size_t)BB * MPTS * 32 * 2);
  unsigned short* BXP   = (unsigned short*)alloc((size_t)BB * MPTS * 32 * 2);
  // packed weights
  unsigned short* Wsf1 = (unsigned short*)alloc(64 * 32 * 2);
  unsigned short* Wsf2 = (unsigned short*)alloc(64 * 64 * 2);
  unsigned short* Wsf3 = (unsigned short*)alloc(64 * 64 * 2);
  unsigned short* Wsf4 = (unsigned short*)alloc(128 * 64 * 2);
  unsigned short* Wsf5 = (unsigned short*)alloc(1024 * 128 * 2);
  unsigned short* Wsh1 = (unsigned short*)alloc(512 * 1120 * 2);
  unsigned short* Wsh2 = (unsigned short*)alloc(256 * 512 * 2);
  unsigned short* Wsh3 = (unsigned short*)alloc(128 * 256 * 2);
  unsigned short* Wsh4 = (unsigned short*)alloc(128 * 128 * 2);
  unsigned short* Wt1  = (unsigned short*)alloc(128 * 32 * 2);
  unsigned short* Wt2  = (unsigned short*)alloc(256 * 128 * 2);
  unsigned short* Wt3  = (unsigned short*)alloc(512 * 256 * 2);
  unsigned short* Wb1  = (unsigned short*)alloc(128 * 32 * 2);
  unsigned short* Wb2  = (unsigned short*)alloc(128 * 128 * 2);
  unsigned short* Wb3  = (unsigned short*)alloc(256 * 128 * 2);
  unsigned short* Wb4  = (unsigned short*)alloc(512 * 256 * 2);
  // smalls
  float* G1    = (float*)alloc(BB * 1024 * 4);
  float* GC    = (float*)alloc(BB * 1027 * 4);
  float* STATS = (float*)alloc(2 * 1024 * 4);
  float* LOG   = (float*)alloc((size_t)BB * 2 * NPTS * 4);
  int*   MBUF  = (int*)  alloc((size_t)BB * NPTS * 4);
  int*   CNT   = (int*)  alloc(BB * 4);
  float* MEANB = (float*)alloc(BB * 3 * 4);
  float* TX    = (float*)alloc((size_t)BB * 3 * MPTS * 4);
  float* GT    = (float*)alloc(BB * 512 * 4);
  float* GTC   = (float*)alloc(BB * 515 * 4);
  float* TH1   = (float*)alloc(BB * 256 * 4);
  float* TH2   = (float*)alloc(BB * 128 * 4);
  float* CD    = (float*)alloc(BB * 3 * 4);
  float* S1C   = (float*)alloc(BB * 3 * 4);
  float* GB    = (float*)alloc(BB * 512 * 4);
  float* GBC   = (float*)alloc(BB * 515 * 4);
  float* BH1   = (float*)alloc(BB * 512 * 4);
  float* BH2   = (float*)alloc(BB * 256 * 4);
  float* BOX   = (float*)alloc(BB * OUTD * 4);
  if (off > ws_size) return;                 // workspace too small: deterministic no-op

  auto packw = [&](const float* W, unsigned short* Wp, int C, int Cp, int O) {
    pack_w<<<(O * Cp + 255) / 256, 256, 0, stream>>>(W, Wp, C, Cp, O);
  };
  // GEMM + BN(+relu) -> packed next-layer input
  auto conv_bn = [&](const unsigned short* Wp, const float* bias, const unsigned short* Xp,
                     unsigned short* OutP, int Cp2, int O, int N, int relu) {
    int wpb = (O >= 128) ? 8 : (O / 16);
    dim3 grd(N / 32, O / (16 * wpb), BB);
    conv_wmma_lds<<<grd, 32 * wpb, 0, stream>>>((const unsigned*)Wp, bias,
                                                (const unsigned*)Xp, YF, Cp2, O, N);
    bn_stats_conv<<<O, 256, 0, stream>>>(YF, STATS, O, N);
    long long tot = (long long)BB * O * N;
    bn_apply_pack<<<(int)((tot + 255) / 256), 256, 0, stream>>>(YF, STATS, OutP, O, N, relu);
  };

  // ---- pack weights ----
  packw(sf_w1, Wsf1, 4, 32, 64);     packw(sf_w2, Wsf2, 64, 64, 64);
  packw(sf_w3, Wsf3, 64, 64, 64);    packw(sf_w4, Wsf4, 64, 64, 128);
  packw(sf_w5, Wsf5, 128, 128, 1024);
  packw(sh_w1, Wsh1, 1091, 1120, 512); packw(sh_w2, Wsh2, 512, 512, 256);
  packw(sh_w3, Wsh3, 256, 256, 128);   packw(sh_w4, Wsh4, 128, 128, 128);
  packw(t_w1, Wt1, 3, 32, 128);  packw(t_w2, Wt2, 128, 128, 256);
  packw(t_w3, Wt3, 256, 256, 512);
  packw(b_w1, Wb1, 3, 32, 128);  packw(b_w2, Wb2, 128, 128, 128);
  packw(b_w3, Wb3, 128, 128, 256); packw(b_w4, Wb4, 256, 256, 512);

  // ---- instance feature net ----
  pack_pc<<<(BB * NPTS + 255) / 256, 256, 0, stream>>>(pc, XP);
  conv_bn(Wsf1, sf_b1, XP,  AP0, 16,  64,   NPTS, 1);
  conv_bn(Wsf2, sf_b2, AP0, PFP, 32,  64,   NPTS, 1);
  conv_bn(Wsf3, sf_b3, PFP, AP0, 32,  64,   NPTS, 1);
  conv_bn(Wsf4, sf_b4, AP0, AP1, 32,  128,  NPTS, 1);
  conv_bn(Wsf5, sf_b5, AP1, AP0, 64,  1024, NPTS, 0);        // no relu
  maxpool_packed<<<(BB * 1024 + 255) / 256, 256, 0, stream>>>(AP0, G1, 1024, NPTS);
  concat_pre<<<(BB * 1027 + 255) / 256, 256, 0, stream>>>(oneh, 3, G1, 1024, GC);
  build_feat<<<(int)(((long long)BB * NPTS * 1120 + 255) / 256), 256, 0, stream>>>(GC, PFP, FEATP);

  // ---- segmentation net ----
  conv_bn(Wsh1, sh_b1, FEATP, AP1, 560, 512, NPTS, 1);
  conv_bn(Wsh2, sh_b2, AP1,   AP0, 256, 256, NPTS, 1);
  conv_bn(Wsh3, sh_b3, AP0,   AP1, 128, 128, NPTS, 1);
  conv_bn(Wsh4, sh_b4, AP1,   AP0, 64,  128, NPTS, 1);
  conv2_packed<<<(int)(((long long)BB * 2 * NPTS + 255) / 256), 256, 0, stream>>>(
      sh_w5, sh_b5, AP0, LOG, 128, 2, NPTS);                 // logits, no BN

  // ---- mask / center / sample ----
  mask_reduce<<<BB, 256, 0, stream>>>(LOG, pc, MBUF, CNT, MEANB);
  sample_gather<<<BB, 256, 0, stream>>>(MBUF, CNT, MEANB, pc, TX, TXP);

  // ---- T-net ----
  conv_bn(Wt1, t_b1, TXP, AP0, 16,  128, MPTS, 1);
  conv_bn(Wt2, t_b2, AP0, AP1, 64,  256, MPTS, 1);
  conv_bn(Wt3, t_b3, AP1, AP0, 128, 512, MPTS, 1);
  maxpool_packed<<<(BB * 512 + 255) / 256, 256, 0, stream>>>(AP0, GT, 512, MPTS);
  concat_pre<<<(BB * 515 + 255) / 256, 256, 0, stream>>>(oneh, 3, GT, 512, GTC);
  fc_kernel<<<(BB * 256 + 255) / 256, 256, 0, stream>>>(GTC, t_fw1, t_fb1, TH1, 515, 256);
  bn_fc<<<1, 256, 0, stream>>>(TH1, 256, 1);
  fc_kernel<<<(BB * 128 + 255) / 256, 256, 0, stream>>>(TH1, t_fw2, t_fb2, TH2, 256, 128);
  bn_fc<<<1, 128, 0, stream>>>(TH2, 128, 1);
  fc_kernel<<<(BB * 3 + 255) / 256, 256, 0, stream>>>(TH2, t_fw3, t_fb3, CD, 128, 3);
  add_center<<<1, 128, 0, stream>>>(CD, MEANB, S1C);
  sub_center_pack<<<(BB * MPTS + 255) / 256, 256, 0, stream>>>(TX, CD, BXP);

  // ---- box net ----
  conv_bn(Wb1, b_b1, BXP, AP0, 16,  128, MPTS, 1);
  conv_bn(Wb2, b_b2, AP0, AP1, 64,  128, MPTS, 1);
  conv_bn(Wb3, b_b3, AP1, AP0, 64,  256, MPTS, 1);
  conv_bn(Wb4, b_b4, AP0, AP1, 128, 512, MPTS, 1);
  maxpool_packed<<<(BB * 512 + 255) / 256, 256, 0, stream>>>(AP1, GB, 512, MPTS);
  concat_pre<<<(BB * 515 + 255) / 256, 256, 0, stream>>>(GB, 512, oneh, 3, GBC);
  fc_kernel<<<(BB * 512 + 255) / 256, 256, 0, stream>>>(GBC, b_fw1, b_fb1, BH1, 515, 512);
  bn_fc<<<2, 256, 0, stream>>>(BH1, 512, 1);
  fc_kernel<<<(BB * 256 + 255) / 256, 256, 0, stream>>>(BH1, b_fw2, b_fb2, BH2, 512, 256);
  bn_fc<<<1, 256, 0, stream>>>(BH2, 256, 1);
  fc_kernel<<<(BB * OUTD + 255) / 256, 256, 0, stream>>>(BH2, b_fw3, b_fb3, BOX, 256, OUTD);

  // ---- final output ----
  assemble_out<<<(int)(((long long)BB * (2 * NPTS + OUTD + 3) + 255) / 256), 256, 0, stream>>>(
      LOG, BOX, S1C, (float*)d_out);
}
